// Barrier_Net_48275432407833
// MI455X (gfx1250) — matrix operand order repaired and checked by reading, not compiled, not run on hardware
//
#include <hip/hip_runtime.h>
#include <hip/hip_bf16.h>

typedef __attribute__((ext_vector_type(16))) _Float16 v16h;
typedef __attribute__((ext_vector_type(8)))  float    v8f;
typedef __attribute__((ext_vector_type(4)))  float    v4f;

#define XDIM 85   // 1 + 4 + 16*4 + 8*2
#define NW   4    // waves per block, each wave owns a 16-row batch tile

#define PHI_STRIDE 72    // floats; 288B rows: 16B-aligned, bank-spread
#define F_STRIDE   132   // floats; 528B rows: 16B-aligned, bank-spread
// sF row layout (per batch row): [0..15] sumphi_n | [16..31] zero pad
//                                [32..47] sumphi_o | [48..63] zero pad
//                                [64..79] rho_n | [80..95] rho_o | [96..99] g | [100..131] zero

static __device__ __forceinline__ v8f wmma_f16(v16h a, v16h b, v8f c) {
  return __builtin_amdgcn_wmma_f32_16x16x32_f16(false, a, false, b, (short)0, c, false, false);
}

// Compiler/memory-model fence only: all LDS buffers are wave-private and DS ops
// from one wave are in-order in hardware, so no workgroup barrier is needed.
static __device__ __forceinline__ void wave_fence() {
  __builtin_amdgcn_fence(__ATOMIC_ACQ_REL, "wavefront");
}

static __device__ __forceinline__ v8f splat8(float b) {
  v8f c;
#pragma unroll
  for (int r = 0; r < 8; ++r) c[r] = b;
  return c;
}

// Single-instruction ReLU: avoids clang's canonicalize (max(v,v)) before maxnum.
static __device__ __forceinline__ v8f relu8(v8f c) {
#pragma unroll
  for (int r = 0; r < 8; ++r) {
    float y;
    asm("v_max_num_f32_e32 %0, 0, %1" : "=v"(y) : "v"(c[r]));
    c[r] = y;
  }
  return c;
}

static __device__ __forceinline__ float sum8(v8f c) {
  return ((c[0] + c[1]) + (c[2] + c[3])) + ((c[4] + c[5]) + (c[6] + c[7]));
}

// 16 f32 -> v16h in WMMA-A register order (elems 0..7 = K kb..kb+7, 8..15 = K 16+kb..)
static __device__ __forceinline__ v16h cvt16(v4f a0, v4f a1, v4f a2, v4f a3) {
  v16h a;
#pragma unroll
  for (int j = 0; j < 4; ++j) {
    a[j]      = (_Float16)a0[j];
    a[4 + j]  = (_Float16)a1[j];
    a[8 + j]  = (_Float16)a2[j];
    a[12 + j] = (_Float16)a3[j];
  }
  return a;
}

// A-tile (16x32, f16) from f32 LDS, row-major [m][k]; K-chunk [k0,k0+32).
// Rows must be 16B-aligned and padding cols pre-zeroed in LDS.
static __device__ __forceinline__ v16h load_A32(const float* S, int stride, int k0, int lane) {
  int m  = lane & 15;
  int kb = (lane & 16) ? 8 : 0;
  const float* p = S + m * stride + k0 + kb;
  return cvt16(*(const v4f*)(p), *(const v4f*)(p + 4),
               *(const v4f*)(p + 16), *(const v4f*)(p + 20));
}

// B-tile (32x16, f16) from row-major f32 W[K][ncols]; branch-free clamp+select.
static __device__ __forceinline__ v16h load_B(const float* W, int ncols,
                                              int n0, int k0, int kmax, int nmax, int lane) {
  int n   = n0 + (lane & 15);
  int kb  = k0 + ((lane & 16) ? 8 : 0);
  bool okn = n < nmax;
  v16h b;
#pragma unroll
  for (int r = 0; r < 8; ++r) {
    int ka  = kb + 2 * r + ((r >= 4) ? 8 : 0);
    bool ok0 = okn && (ka < kmax);
    bool ok1 = okn && (ka + 1 < kmax);
    float f0 = W[ok0 ? ka * ncols + n : 0];        // unconditional load, clamped addr
    float f1 = W[ok1 ? (ka + 1) * ncols + n : 0];
    b[2 * r]     = (_Float16)(ok0 ? f0 : 0.f);
    b[2 * r + 1] = (_Float16)(ok1 ? f1 : 0.f);
  }
  return b;
}

// Scatter 16x16 f32 C-tile (VGPR r, lane l -> M=r+8*(l>=16), N=l%16) into LDS.
static __device__ __forceinline__ void store_C(float* S, int stride, v8f c, int n0, int lane) {
  int n  = n0 + (lane & 15);
  int m0 = (lane & 16) ? 8 : 0;
#pragma unroll
  for (int r = 0; r < 8; ++r) S[(m0 + r) * stride + n] = c[r];
}

__global__ __launch_bounds__(NW * 32)
void barrier_net_kernel(const float* __restrict__ x,
    const float* __restrict__ Wp1n, const float* __restrict__ bp1n,
    const float* __restrict__ Wp2n, const float* __restrict__ bp2n,
    const float* __restrict__ Wr1n, const float* __restrict__ br1n,
    const float* __restrict__ Wr2n, const float* __restrict__ br2n,
    const float* __restrict__ Wp1o, const float* __restrict__ bp1o,
    const float* __restrict__ Wp2o, const float* __restrict__ bp2o,
    const float* __restrict__ Wr1o, const float* __restrict__ br1o,
    const float* __restrict__ Wr2o, const float* __restrict__ br2o,
    const float* __restrict__ Wpsi1, const float* __restrict__ bpsi1,
    const float* __restrict__ Wpsi2, const float* __restrict__ bpsi2,
    float* __restrict__ out)
{
  __shared__ __align__(16) float sX[NW][16 * XDIM + 16];     // staged input tile
  __shared__ __align__(16) float sPhi[NW][16 * PHI_STRIDE];  // layer-1 activations
  __shared__ __align__(16) float sF[NW][16 * F_STRIDE];      // feature rows (see layout)
  __shared__ float sE[NW][16][2];                            // psi outputs

  const int lane = threadIdx.x & 31;
  const int wave = threadIdx.x >> 5;
  const int nl   = lane & 15;
  const int tile0 = (blockIdx.x * NW + wave) * 16;

  float* sXw = &sX[wave][0];
  float* phi = &sPhi[wave][0];
  float* F   = &sF[wave][0];
  const float* zsrc = F + 100;   // permanently-zero LDS words (psi pad region, row 0)

  // ---- stage input rows (coalesced) and zero the feature buffer ----
  {
    const float* xg = x + (size_t)tile0 * XDIM;
#pragma unroll
    for (int t = 0; t < (16 * XDIM + 31) / 32; ++t) {
      int idx = t * 32 + lane;
      if (idx < 16 * XDIM) sXw[idx] = xg[idx];
    }
    v4f z4 = {};
#pragma unroll
    for (int t = 0; t < (16 * F_STRIDE / 4 + 31) / 32; ++t) {
      int idx = t * 32 + lane;
      if (idx < 16 * F_STRIDE / 4) ((v4f*)F)[idx] = z4;
    }
  }
  wave_fence();
  // stage goal vectors g into psi region cols 96..99 (64 values, 2 full-wave steps)
#pragma unroll
  for (int t = 0; t < 2; ++t) {
    int idx = t * 32 + lane;   // 0..63
    F[(idx >> 2) * F_STRIDE + 96 + (idx & 3)] = sXw[(idx >> 2) * XDIM + 1 + (idx & 3)];
  }

  // ---- hoist phi-network weights (B-tiles) and biases out of the row loop ----
  v16h Bp1n_t[4], Bp1o_t[4], Bp2n_t[2], Bp2o_t[2];
#pragma unroll
  for (int nc = 0; nc < 4; ++nc) {
    Bp1n_t[nc] = load_B(Wp1n, 64, nc * 16, 0, 4, 64, lane);
    Bp1o_t[nc] = load_B(Wp1o, 64, nc * 16, 0, 2, 64, lane);
  }
#pragma unroll
  for (int kc = 0; kc < 2; ++kc) {
    Bp2n_t[kc] = load_B(Wp2n, 16, 0, kc * 32, 64, 16, lane);
    Bp2o_t[kc] = load_B(Wp2o, 16, 0, kc * 32, 64, 16, lane);
  }
  float b1n[4], b1o[4];
#pragma unroll
  for (int nc = 0; nc < 4; ++nc) { b1n[nc] = bp1n[nc * 16 + nl]; b1o[nc] = bp1o[nc * 16 + nl]; }
  const float b2n = bp2n[nl], b2o = bp2o[nl];
  wave_fence();

  // ======== Phase A: per-row phi_n; obstacles packed two rows per tile ========
#pragma unroll 1
  for (int ip = 0; ip < 8; ++ip) {
    const int i0 = ip * 2;

    // --- neighbors for rows i0, i0+1 (rolled: keeps VGPR pressure < 256) ---
#pragma unroll 1
    for (int half = 0; half < 2; ++half) {
      const int i = i0 + half;
      const float* sxr = sXw + i * XDIM;
      v16h An = {};
      {
        int m = lane & 15;
        const float* p = (lane < 16) ? (sxr + 5 + 4 * m) : zsrc;  // addr select, no elem masks
        An[0] = (_Float16)p[0];
        An[1] = (_Float16)p[1];
        An[2] = (_Float16)p[2];
        An[3] = (_Float16)p[3];
      }
#pragma unroll
      for (int nc = 0; nc < 4; ++nc) {
        v8f c = wmma_f16(An, Bp1n_t[nc], splat8(b1n[nc]));  // bias folded into C
        store_C(phi, PHI_STRIDE, relu8(c), nc * 16, lane);
      }
      wave_fence();
      v8f p2 = {};
      p2 = wmma_f16(load_A32(phi, PHI_STRIDE, 0, lane),  Bp2n_t[0], p2);
      p2 = wmma_f16(load_A32(phi, PHI_STRIDE, 32, lane), Bp2n_t[1], p2);
      float s = sum8(p2);
      s += __shfl_xor(s, 16, 32);     // combine M=0..7 with M=8..15
      F[i * F_STRIDE + nl] = s + 16.f * b2n;  // both half-waves write identical value
      wave_fence();
    }

    // --- obstacles, two rows per tile: rows 0-7 = i0, rows 8-15 = i0+1 ---
    v16h Ao = {};
    {
      int m = lane & 15;
      const float* p = (lane < 16)
          ? (sXw + (i0 + (m >> 3)) * XDIM + 69 + 2 * (m & 7))
          : zsrc;
      Ao[0] = (_Float16)p[0];
      Ao[1] = (_Float16)p[1];
    }
#pragma unroll
    for (int nc = 0; nc < 4; ++nc) {
      v8f c = wmma_f16(Ao, Bp1o_t[nc], splat8(b1o[nc]));
      store_C(phi, PHI_STRIDE, relu8(c), nc * 16, lane);
    }
    wave_fence();
    v8f q = {};
    q = wmma_f16(load_A32(phi, PHI_STRIDE, 0, lane),  Bp2o_t[0], q);
    q = wmma_f16(load_A32(phi, PHI_STRIDE, 32, lane), Bp2o_t[1], q);
    // lanes 0-15 hold row i0 sums (M=0..7); lanes 16-31 hold row i0+1 sums (M=8..15)
    float so = sum8(q) + 8.f * b2o;
    F[(i0 + (lane >> 4)) * F_STRIDE + 32 + nl] = so;
    wave_fence();
  }

  // ================= Phase B: rho / psi over the 16-row tile =================
  v16h ArN = load_A32(F, F_STRIDE, 0, lane);        // sumphi_n | zero pad
  v16h ArO = load_A32(F + 32, F_STRIDE, 0, lane);   // sumphi_o | zero pad

  // rho_n
  v8f rn;
  {
#pragma unroll
    for (int nc = 0; nc < 4; ++nc) {
      v8f c = wmma_f16(ArN, load_B(Wr1n, 64, nc * 16, 0, 16, 64, lane),
                       splat8(br1n[nc * 16 + nl]));
      store_C(phi, PHI_STRIDE, relu8(c), nc * 16, lane);
    }
    wave_fence();
    v8f c = splat8(br2n[nl]);
    c = wmma_f16(load_A32(phi, PHI_STRIDE, 0, lane),  load_B(Wr2n, 16, 0, 0, 64, 16, lane), c);
    c = wmma_f16(load_A32(phi, PHI_STRIDE, 32, lane), load_B(Wr2n, 16, 0, 32, 64, 16, lane), c);
    rn = c;
  }
  wave_fence();

  // rho_o
  v8f ro;
  {
#pragma unroll
    for (int nc = 0; nc < 4; ++nc) {
      v8f c = wmma_f16(ArO, load_B(Wr1o, 64, nc * 16, 0, 16, 64, lane),
                       splat8(br1o[nc * 16 + nl]));
      store_C(phi, PHI_STRIDE, relu8(c), nc * 16, lane);
    }
    wave_fence();
    v8f c = splat8(br2o[nl]);
    c = wmma_f16(load_A32(phi, PHI_STRIDE, 0, lane),  load_B(Wr2o, 16, 0, 0, 64, 16, lane), c);
    c = wmma_f16(load_A32(phi, PHI_STRIDE, 32, lane), load_B(Wr2o, 16, 0, 32, 64, 16, lane), c);
    ro = c;
  }
  wave_fence();

  // h = [rho_n | rho_o | g | 0pad] into psi region (cols 64..131)
  store_C(F, F_STRIDE, rn, 64, lane);
  store_C(F, F_STRIDE, ro, 80, lane);
  wave_fence();

  // psi layer 1: (16 x 36) @ (36 x 64), K padded to 64 via zeroed LDS cols
  v16h Ah0 = load_A32(F + 64, F_STRIDE, 0, lane);
  v16h Ah1 = load_A32(F + 64, F_STRIDE, 32, lane);
#pragma unroll
  for (int nc = 0; nc < 4; ++nc) {
    v8f c = splat8(bpsi1[nc * 16 + nl]);
    c = wmma_f16(Ah0, load_B(Wpsi1, 64, nc * 16, 0, 36, 64, lane), c);
    c = wmma_f16(Ah1, load_B(Wpsi1, 64, nc * 16, 32, 36, 64, lane), c);
    store_C(phi, PHI_STRIDE, relu8(c), nc * 16, lane);
  }
  wave_fence();

  // psi layer 2: (16 x 64) @ (64 x 2), N padded to 16; bias folded into C
  v8f e = splat8((nl < 2) ? bpsi2[nl & 1] : 0.f);
  e = wmma_f16(load_A32(phi, PHI_STRIDE, 0, lane),  load_B(Wpsi2, 2, 0, 0, 64, 2, lane), e);
  e = wmma_f16(load_A32(phi, PHI_STRIDE, 32, lane), load_B(Wpsi2, 2, 0, 32, 64, 2, lane), e);
  {
    int m0 = (lane & 16) ? 8 : 0;
    if (nl < 2) {
#pragma unroll
      for (int r = 0; r < 8; ++r) sE[wave][m0 + r][nl] = e[r];
    }
  }
  wave_fence();

  // ================= Epilogue: tanh squash, barrier sums, normalize =========
  if (lane < 16) {
    int row = tile0 + lane;
    const float* sxr = sXw + lane * XDIM;
    // (tanh(e)+1)/2 * (PHI_MAX-PHI_MIN) + PHI_MIN == 2*tanh(e)
    float a0 = 2.f * tanhf(sE[wave][lane][0]);
    float a1 = 2.f * tanhf(sE[wave][lane][1]);
    float bx = 0.f, by = 0.f;
#pragma unroll
    for (int j = 0; j < 16; ++j) {           // robot barrier, D=0.3
      float px = -sxr[5 + 4 * j], py = -sxr[5 + 4 * j + 1];
      float nrm = sqrtf(px * px + py * py);
      float cc = 0.05f / (nrm * (nrm - 0.3f));
      bx += cc * px; by += cc * py;
    }
#pragma unroll
    for (int j = 0; j < 8; ++j) {            // obstacle barrier, D=0.3
      float px = -sxr[69 + 2 * j], py = -sxr[69 + 2 * j + 1];
      float nrm = sqrtf(px * px + py * py);
      float cc = 0.05f / (nrm * (nrm - 0.3f));
      bx += cc * px; by += cc * py;
    }
    a0 += bx; a1 += by;
    float inv = fmaxf(fmaxf(fabsf(a0), fabsf(a1)) * 0.5f, 1.f);  // /A_MAX
    out[2 * row]     = a0 / inv;
    out[2 * row + 1] = a1 / inv;
  }
}

extern "C" void kernel_launch(void* const* d_in, const int* in_sizes, int n_in,
                              void* d_out, int out_size, void* d_ws, size_t ws_size,
                              hipStream_t stream) {
  const float* x     = (const float*)d_in[0];
  const float* Wp1n  = (const float*)d_in[1];
  const float* bp1n  = (const float*)d_in[2];
  const float* Wp2n  = (const float*)d_in[3];
  const float* bp2n  = (const float*)d_in[4];
  const float* Wr1n  = (const float*)d_in[5];
  const float* br1n  = (const float*)d_in[6];
  const float* Wr2n  = (const float*)d_in[7];
  const float* br2n  = (const float*)d_in[8];
  const float* Wp1o  = (const float*)d_in[9];
  const float* bp1o  = (const float*)d_in[10];
  const float* Wp2o  = (const float*)d_in[11];
  const float* bp2o  = (const float*)d_in[12];
  const float* Wr1o  = (const float*)d_in[13];
  const float* br1o  = (const float*)d_in[14];
  const float* Wr2o  = (const float*)d_in[15];
  const float* br2o  = (const float*)d_in[16];
  const float* Wpsi1 = (const float*)d_in[17];
  const float* bpsi1 = (const float*)d_in[18];
  const float* Wpsi2 = (const float*)d_in[19];
  const float* bpsi2 = (const float*)d_in[20];

  int nrows  = in_sizes[0] / XDIM;          // 131072
  int blocks = nrows / (16 * NW);           // 2048 blocks of 4 waves

  barrier_net_kernel<<<blocks, NW * 32, 0, stream>>>(
      x, Wp1n, bp1n, Wp2n, bp2n, Wr1n, br1n, Wr2n, br2n,
      Wp1o, bp1o, Wp2o, bp2o, Wr1o, br1o, Wr2o, br2o,
      Wpsi1, bpsi1, Wpsi2, bpsi2, (float*)d_out);
}
